// CG_19628000542733
// MI455X (gfx1250) — compile-verified, matrix operand order checked by reading, not run on hardware
//
#include <hip/hip_runtime.h>
#include <hip/hip_bf16.h>

// ---------------------------------------------------------------------------
// GraphMAE-style pipeline for MI455X (gfx1250, wave32).
// D = H = 256 hardcoded (matches reference). L = 3 layers.
// GEMMs use v_wmma_f32_16x16x32_bf16 (f32 accumulate), fully unrolled K loop.
// ---------------------------------------------------------------------------

#define KDIM 256
#define NOUT 256
#define BN_EPS 1e-5f

typedef __attribute__((ext_vector_type(16))) __bf16 v16bf;
typedef __attribute__((ext_vector_type(8)))  float  v8f;

__device__ __forceinline__ void atomic_add_f32(float* p, float v) {
    unsafeAtomicAdd(p, v);   // lowers to global_atomic_add_f32 on gfx1250
}

// pack 4 floats -> 4 bf16 (two v_cvt_pk_bf16_f32) as uint2 for ds_store_b64
__device__ __forceinline__ uint2 pack4_bf16(float4 v) {
    union { __bf16 h[4]; uint2 u; } t;
    t.h[0] = (__bf16)v.x; t.h[1] = (__bf16)v.y;
    t.h[2] = (__bf16)v.z; t.h[3] = (__bf16)v.w;
    return t.u;
}

// ---------------------------------------------------------------------------
// GEMM: Y[M x 256] = X[M x 256] * W^T, W is [256 x 256] Linear weight [out,in]
// Block = 256 threads = 8 waves. Block tile 32(M) x 64(N); wave tile 16x16.
// bf16 operands staged in LDS (36-elem padded rows: 8B-aligned float4 groups,
// conflict-free 72B row stride), f32 accumulation via WMMA.
// ---------------------------------------------------------------------------
__global__ __launch_bounds__(256) void k_gemm_nt(
    const float* __restrict__ X,
    const float* __restrict__ W,
    float* __restrict__ Y,
    int M)
{
    __shared__ __bf16 At[32][36];
    __shared__ __bf16 Bt[64][36];

    const int tid  = threadIdx.x;
    const int lane = tid & 31;
    const int wave = tid >> 5;
    const int wrow = wave & 1;      // 2 wave-rows  -> 32 M
    const int wcol = wave >> 1;     // 4 wave-cols  -> 64 N
    const int lh   = lane >> 4;     // half-wave selects K group
    const int mn   = lane & 15;     // row (A) / col (B) within 16

    const int mBase = blockIdx.x * 32;
    const int nBase = blockIdx.y * 64;

    // cooperative-load coordinates: one float4 (A) / two float4 (B) per thread
    const int arow   = tid >> 3;           // 0..31
    const int acol   = (tid & 7) * 4;      // 0,4,...,28
    const int garow  = mBase + arow;
    const int garowc = min(garow, M - 1);  // clamped: loads always legal
    const bool avalid = garow < M;

    v8f acc = {};

    #pragma unroll
    for (int kc = 0; kc < KDIM; kc += 32) {
        // A tile 32x32: vectorized B128 load, branchless zero for padded rows
        float4 va = *(const float4*)(X + (size_t)garowc * KDIM + kc + acol);
        if (!avalid) { va.x = 0.f; va.y = 0.f; va.z = 0.f; va.w = 0.f; }
        *(uint2*)&At[arow][acol] = pack4_bf16(va);

        // B tile 64x32 (rows = output channels; always in range)
        #pragma unroll
        for (int j = 0; j < 2; ++j) {
            int idx = tid + j * 256;
            int r = idx >> 3, c = (idx & 7) * 4;
            float4 vb = *(const float4*)(W + (size_t)(nBase + r) * KDIM + kc + c);
            *(uint2*)&Bt[r][c] = pack4_bf16(vb);
        }
        __syncthreads();

        // fragment load per CDNA5 16-bit A 16x32 layout:
        // lanes 0-15: VGPR p<4 -> K=2p..2p+1, p>=4 -> K=16+2(p-4)..; lanes16-31: +8
        v16bf a, b;
        #pragma unroll
        for (int e = 0; e < 16; ++e) {
            int p = e >> 1;
            int k = ((p < 4) ? (lh * 8 + p * 2) : (16 + lh * 8 + (p - 4) * 2)) + (e & 1);
            a[e] = At[wrow * 16 + mn][k];
            b[e] = Bt[wcol * 16 + mn][k];
        }
        acc = __builtin_amdgcn_wmma_f32_16x16x32_bf16(
            false, a, false, b, (short)0, acc, false, false);
        __syncthreads();
    }

    // C/D layout: VGPR v -> M = v + 8*half; N = lane&15
    const int col = nBase + wcol * 16 + mn;
    #pragma unroll
    for (int v = 0; v < 8; ++v) {
        int row = mBase + wrow * 16 + v + 8 * lh;
        if (row < M) Y[(size_t)row * NOUT + col] = acc[v];
    }
}

// ---------------------------------------------------------------------------
// GIN aggregation: agg[dst] += h[src], one edge per wave (256 floats/row).
// h and agg are 51 MB each -> L2 resident on MI455X (192 MB L2).
// ---------------------------------------------------------------------------
__global__ __launch_bounds__(256) void k_edge_scatter(
    const float* __restrict__ h, float* __restrict__ agg,
    const int* __restrict__ src, const int* __restrict__ dst, int E)
{
    int gw   = (int)((blockIdx.x * 256 + threadIdx.x) >> 5);
    int lane = threadIdx.x & 31;
    if (gw >= E) return;
    int s = src[gw], d = dst[gw];
    const float4* hs = (const float4*)(h + (size_t)s * 256);
    float* ad = agg + (size_t)d * 256;
    #pragma unroll
    for (int j = 0; j < 2; ++j) {
        float4 v = hs[lane + 32 * j];
        int base = (lane + 32 * j) * 4;
        atomic_add_f32(ad + base + 0, v.x);
        atomic_add_f32(ad + base + 1, v.y);
        atomic_add_f32(ad + base + 2, v.z);
        atomic_add_f32(ad + base + 3, v.w);
    }
}

// ---------------------------------------------------------------------------
// BatchNorm (training mode, batch stats): per-column sum / sumsq
// ---------------------------------------------------------------------------
__global__ __launch_bounds__(256) void k_bn_stats(
    const float* __restrict__ X, int M, float* __restrict__ sums)
{
    int c  = threadIdx.x;                // column (coalesced across threads)
    int r0 = blockIdx.x * 256;
    int r1 = min(r0 + 256, M);
    float s = 0.f, s2 = 0.f;
    for (int r = r0; r < r1; ++r) {
        float v = X[(size_t)r * 256 + c];
        s += v; s2 += v * v;
    }
    atomic_add_f32(&sums[c], s);
    atomic_add_f32(&sums[256 + c], s2);
}

__global__ void k_bn_finalize(
    const float* __restrict__ sums,
    const float* __restrict__ g, const float* __restrict__ b,
    float invM, float* __restrict__ ss)
{
    int c = threadIdx.x;
    float mean = sums[c] * invM;
    float var  = sums[256 + c] * invM - mean * mean;   // biased variance
    float sc   = g[c] * rsqrtf(var + BN_EPS);
    ss[c]       = sc;
    ss[256 + c] = b[c] - mean * sc;
}

// BN-apply + ReLU; optional dual-write (serves as the "agg = copy(h)" pass)
__global__ __launch_bounds__(256) void k_bn_apply_relu(
    float* __restrict__ X, float* __restrict__ dup,
    const float* __restrict__ ss, int M)
{
    int row = blockIdx.x;
    if (row >= M) return;
    int c = threadIdx.x;
    size_t idx = (size_t)row * 256 + c;
    float v = fmaxf(X[idx] * ss[c] + ss[256 + c], 0.0f);
    X[idx] = v;
    if (dup) dup[idx] = v;
}

// ---------------------------------------------------------------------------
// Init / mask / gather / zero
// ---------------------------------------------------------------------------
__global__ void k_copy_dual(const float* __restrict__ s,
                            float* __restrict__ a, float* __restrict__ b, size_t n)
{
    size_t i = (size_t)blockIdx.x * blockDim.x + threadIdx.x;
    if (i < n) { float v = s[i]; a[i] = v; b[i] = v; }
}

__global__ void k_mask_rows(float* __restrict__ a, float* __restrict__ b,
                            const int* __restrict__ ring,
                            const float* __restrict__ tok, int R)
{
    int i = blockIdx.x;
    if (i >= R) return;
    int r = ring[i];
    float t = tok[threadIdx.x];
    a[(size_t)r * 256 + threadIdx.x] = t;
    b[(size_t)r * 256 + threadIdx.x] = t;
}

__global__ void k_gather_dual(const float* __restrict__ feat,
                              const int* __restrict__ ring,
                              float* __restrict__ a, float* __restrict__ b, int R)
{
    int i = blockIdx.x;
    if (i >= R) return;
    int r = ring[i];
    float v = feat[(size_t)r * 256 + threadIdx.x];
    a[(size_t)i * 256 + threadIdx.x] = v;
    b[(size_t)i * 256 + threadIdx.x] = v;
}

__global__ void k_zero(float* __restrict__ p, int n)
{
    int i = blockIdx.x * 256 + threadIdx.x;
    if (i < n) p[i] = 0.f;
}

// ---------------------------------------------------------------------------
// Loss: mean(1 - cos(h1[ring[i]], h2[i])), one row per wave
// ---------------------------------------------------------------------------
__global__ __launch_bounds__(256) void k_loss(
    const float* __restrict__ h1, const int* __restrict__ ring,
    const float* __restrict__ h2, int R, float* __restrict__ accum)
{
    int wave = threadIdx.x >> 5;
    int lane = threadIdx.x & 31;
    int i = blockIdx.x * 8 + wave;
    if (i >= R) return;
    int r = ring[i];
    const float4* x4 = (const float4*)(h1 + (size_t)r * 256);
    const float4* y4 = (const float4*)(h2 + (size_t)i * 256);
    float dot = 0.f, nx = 0.f, ny = 0.f;
    #pragma unroll
    for (int j = 0; j < 2; ++j) {
        float4 x = x4[lane + 32 * j];
        float4 y = y4[lane + 32 * j];
        dot += x.x*y.x + x.y*y.y + x.z*y.z + x.w*y.w;
        nx  += x.x*x.x + x.y*x.y + x.z*x.z + x.w*x.w;
        ny  += y.x*y.x + y.y*y.y + y.z*y.z + y.w*y.w;
    }
    #pragma unroll
    for (int off = 16; off > 0; off >>= 1) {
        dot += __shfl_xor(dot, off, 32);
        nx  += __shfl_xor(nx,  off, 32);
        ny  += __shfl_xor(ny,  off, 32);
    }
    if (lane == 0) {
        float denom = fmaxf(sqrtf(nx), 1e-12f) * fmaxf(sqrtf(ny), 1e-12f);
        atomic_add_f32(accum, 1.0f - dot / denom);
    }
}

__global__ void k_loss_fin(const float* __restrict__ accum,
                           float* __restrict__ out, float invR)
{
    out[0] = accum[0] * invR;
}

// ---------------------------------------------------------------------------
// Host-side orchestration
// ---------------------------------------------------------------------------
extern "C" void kernel_launch(void* const* d_in, const int* in_sizes, int n_in,
                              void* d_out, int out_size, void* d_ws, size_t ws_size,
                              hipStream_t stream)
{
    const float* feat  = (const float*)d_in[0];
    const float* token = (const float*)d_in[1];
    const int*   src   = (const int*)d_in[2];
    const int*   dst   = (const int*)d_in[3];
    const int*   ring  = (const int*)d_in[4];
    const int*   ssrc  = (const int*)d_in[5];
    const int*   sdst  = (const int*)d_in[6];
    const float* onW1  = (const float*)d_in[7];
    const float* onW2  = (const float*)d_in[8];
    const float* ong1  = (const float*)d_in[9];
    const float* onb1  = (const float*)d_in[10];
    const float* ong2  = (const float*)d_in[11];
    const float* onb2  = (const float*)d_in[12];
    const float* tgW1  = (const float*)d_in[13];
    const float* tgW2  = (const float*)d_in[14];
    const float* tgg1  = (const float*)d_in[15];
    const float* tgb1  = (const float*)d_in[16];
    const float* tgg2  = (const float*)d_in[17];
    const float* tgb2  = (const float*)d_in[18];

    const int N  = in_sizes[0] / 256;
    const int E  = in_sizes[2];
    const int R  = in_sizes[4];
    const int Es = in_sizes[5];

    // carve workspace
    float* ws   = (float*)d_ws;
    float* h1   = ws; ws += (size_t)N * 256;
    float* agg1 = ws; ws += (size_t)N * 256;
    float* Y    = ws; ws += (size_t)N * 256;   // shared GEMM temp
    float* h2   = ws; ws += (size_t)R * 256;
    float* agg2 = ws; ws += (size_t)R * 256;
    float* sums = ws; ws += 512;
    float* ss   = ws; ws += 512;
    float* acc  = ws; ws += 1;

    auto run_encoder = [&](float* h, float* agg, int M,
                           const int* es, const int* ed, int ne,
                           const float* W1, const float* W2,
                           const float* g1, const float* b1,
                           const float* g2, const float* b2) {
        const float invM = 1.0f / (float)M;
        dim3 gg((unsigned)((M + 31) / 32), 4);
        for (int l = 0; l < 3; ++l) {
            const float* W1l = W1 + (size_t)l * 256 * 256;
            const float* W2l = W2 + (size_t)l * 256 * 256;
            // agg = h + sum_{e: dst=v} h[src]   (agg pre-initialized to h)
            if (ne > 0)
                k_edge_scatter<<<(ne + 7) / 8, 256, 0, stream>>>(h, agg, es, ed, ne);
            // inner linear + BN + ReLU
            k_gemm_nt<<<gg, 256, 0, stream>>>(agg, W1l, Y, M);
            k_zero<<<2, 256, 0, stream>>>(sums, 512);
            k_bn_stats<<<(M + 255) / 256, 256, 0, stream>>>(Y, M, sums);
            k_bn_finalize<<<1, 256, 0, stream>>>(sums, g1 + l * 256, b1 + l * 256, invM, ss);
            k_bn_apply_relu<<<M, 256, 0, stream>>>(Y, nullptr, ss, M);
            // outer linear + BN + ReLU; dual-write produces next layer's (h, agg)
            k_gemm_nt<<<gg, 256, 0, stream>>>(Y, W2l, agg, M);
            k_zero<<<2, 256, 0, stream>>>(sums, 512);
            k_bn_stats<<<(M + 255) / 256, 256, 0, stream>>>(agg, M, sums);
            k_bn_finalize<<<1, 256, 0, stream>>>(sums, g2 + l * 256, b2 + l * 256, invM, ss);
            k_bn_apply_relu<<<M, 256, 0, stream>>>(agg, h, ss, M);
        }
    };

    // encoder 1: masked features on full graph
    size_t nelem = (size_t)N * 256;
    k_copy_dual<<<(unsigned)((nelem + 255) / 256), 256, 0, stream>>>(feat, h1, agg1, nelem);
    k_mask_rows<<<R, 256, 0, stream>>>(h1, agg1, ring, token, R);
    run_encoder(h1, agg1, N, src, dst, E, onW1, onW2, ong1, onb1, ong2, onb2);

    // encoder 2: original features on induced subgraph
    k_gather_dual<<<R, 256, 0, stream>>>(feat, ring, h2, agg2, R);
    run_encoder(h2, agg2, R, ssrc, sdst, Es, tgW1, tgW2, tgg1, tgb1, tgg2, tgb2);

    // loss
    k_zero<<<1, 256, 0, stream>>>(acc, 1);
    k_loss<<<(R + 7) / 8, 256, 0, stream>>>(h1, ring, h2, R, acc);
    k_loss_fin<<<1, 1, 0, stream>>>(acc, (float*)d_out, 1.0f / (float)R);
}
